// Net_12481174962492
// MI455X (gfx1250) — matrix operand order repaired and checked by reading, not compiled, not run on hardware
//
#include <hip/hip_runtime.h>

// ---------------------------------------------------------------------------
// Types for CDNA5 WMMA (wave32): v16bf A/B fragments, v8f accumulator
// ---------------------------------------------------------------------------
typedef __attribute__((ext_vector_type(16))) __bf16 bf16x16;
typedef __attribute__((ext_vector_type(8)))  float  f32x8;

#define TPB 256
#define NPIX 128
#define BN_EPS 1e-5f

__device__ __forceinline__ __bf16 f2bf(float f) { return (__bf16)f; }

// A-fragment K permutation (ISA 16-bit A 16x32 layout):
// lanes 0-15 hold K 0..7 (v0-3) and 16..23 (v4-7); lanes 16-31 hold 8..15 / 24..31
__device__ __forceinline__ int ka_perm(int half, int i) {
    int vv = i >> 1, j = i & 1;
    return (vv < 4) ? ((half ? 8 : 0) + 2 * vv + j)
                    : ((half ? 24 : 16) + 2 * (vv - 4) + j);
}

// ---------------------------------------------------------------------------
// Implicit-GEMM conv via V_WMMA_F32_16X16X32_BF16.
// Block = 256 threads = 8 wave32. Each block: one output row y (outW==128),
// one tile of 16 output channels, one batch image. Each wave owns 16 pixels.
// K = Cin*KH*KW in chunks of 32. LDS tiles are stored k-contiguous in
// fragment order so each wave fragment is two contiguous ds_load_b128.
//
// Per-chunk im2col decomposition (gk -> ci,ky,kx -> row base, validity) is
// computed ONCE per lane (lane <-> kk) and broadcast with v_readlane during
// staging; it is software-pipelined one chunk ahead, which also provides the
// address stream for global_prefetch_b8. All global indexing inside the hot
// loop is unsigned 32-bit so loads use SGPR-base + 32-bit voffset addressing.
// ---------------------------------------------------------------------------
template <int KH, int KW, int STRIDE, int PAD>
__global__ __launch_bounds__(TPB)
void wmma_conv_kernel(const float* __restrict__ in, const float* __restrict__ w,
                      const float* __restrict__ bias,
                      const float* __restrict__ bnG, const float* __restrict__ bnB,
                      float* __restrict__ out,
                      int Cin, int inH, int inW, int Cout,
                      int relu, int accum)
{
    constexpr int KHW = KH * KW;

    // A: [m][half*16+i] (fragment order, k contiguous)   16*32 bf16 = 1 KB
    // B: [x][k]         (column-major: fragment order)  128*32 bf16 = 8 KB
    __shared__ __align__(16) __bf16 At[16 * 32];
    __shared__ __align__(16) __bf16 Bt[NPIX * 32];

    const int y      = blockIdx.x;               // output row (outH == 128)
    const int ocTile = blockIdx.y;
    const int b      = blockIdx.z;
    const int tid    = threadIdx.x;
    const int lane   = tid & 31;
    const int wave   = tid >> 5;                 // 0..7
    const int half   = lane >> 4;                // lane group 0..15 / 16..31
    const int l16    = lane & 15;
    const int xo     = wave * 16 + l16;          // this lane's output column
    const int xb     = tid & 127;                // staging column for this thread
    const int t0u    = __builtin_amdgcn_readfirstlane(tid >> 7); // wave-uniform

    const int Ktot    = Cin * KHW;
    const int nChunks = (Ktot + 31) >> 5;
    const int ocBase  = ocTile * 16;

    const float* __restrict__ inB = in + (size_t)b * Cin * inH * inW;

    // ---- per-lane decomposition of kk = lane for one chunk ----
    auto decomp = [&](int gk, int& base, int& kx, int& ok) {
        int ci = gk / KHW;                       // constant divisor (templated)
        int r  = gk - ci * KHW;
        int ky = r / KW;
        kx = r - ky * KW;
        int iy = y * STRIDE + ky - PAD;
        ok = (gk < Ktot) & ((unsigned)iy < (unsigned)inH);
        base = (ci * inH + iy) * inW;
    };
    int baseL, kxL, okL;
    decomp(lane, baseL, kxL, okL);

    f32x8 acc = {};

    for (int c = 0; c < nChunks; ++c) {
        const int k0 = c << 5;

        // ---- stage A tile: 512 elems, 2 per thread, k-contiguous + branchless ----
        {
            int idx = tid * 2;
#pragma unroll
            for (int e = 0; e < 2; ++e, ++idx) {
                int m  = idx >> 5;
                int s  = idx & 31;
                int gk = k0 + s;
                int oc = ocBase + m;
                bool ok = (oc < Cout) & (gk < Ktot);
                unsigned off = ok ? (unsigned)(oc * Ktot + gk) : 0u;
                float v = w[off];
                At[idx] = f2bf(ok ? v : 0.0f);
            }
        }
        // ---- stage B tile: 4096 elems (im2col), 16/thread, readlane bcast ----
        {
            const int xs = xb * STRIDE - PAD;
#pragma unroll
            for (int e = 0; e < 16; ++e) {
                int kk   = 2 * e + t0u;          // wave-uniform
                int base = __builtin_amdgcn_readlane(baseL, kk);
                int kx   = __builtin_amdgcn_readlane(kxL, kk);
                int okk  = __builtin_amdgcn_readlane(okL, kk);
                int ix   = xs + kx;
                bool ok  = okk & ((unsigned)ix < (unsigned)inW);
                unsigned off = ok ? (unsigned)(base + ix) : 0u;
                float v  = inB[off];
                Bt[xb * 32 + kk] = f2bf(ok ? v : 0.0f);
            }
        }

        // ---- pipeline: decompose NEXT chunk per lane + prefetch its tile ----
        {
            int baseN, kxN, okN;
            decomp(k0 + 32 + lane, baseN, kxN, okN);
            int ixp = (wave * 16) * STRIDE - PAD + kxN;
            if ((c + 1 < nChunks) & okN & ((unsigned)ixp < (unsigned)inW))
                __builtin_prefetch(&inB[(unsigned)(baseN + ixp)], 0, 1);
            baseL = baseN; kxL = kxN; okL = okN;
        }
        __syncthreads();

        // ---- fragments: 2x ds_load_b128 each (tiles are in fragment order) ----
        union { uint4 q[2]; bf16x16 v; } fa, fb;
        const uint4* Ap = reinterpret_cast<const uint4*>(At);
        const uint4* Bp = reinterpret_cast<const uint4*>(Bt);
        fa.q[0] = Ap[l16 * 4 + half * 2 + 0];
        fa.q[1] = Ap[l16 * 4 + half * 2 + 1];
        fb.q[0] = Bp[xo * 4 + half * 2 + 0];
        fb.q[1] = Bp[xo * 4 + half * 2 + 1];

        acc = __builtin_amdgcn_wmma_f32_16x16x32_bf16(
                  false, fa.v, false, fb.v, (short)0, acc, false, false);
        __syncthreads();
    }

    // ---- epilogue: C/D layout VGPR r -> M = r + 8*half, N = lane%16 ----
    const float bnS = bnG ? rsqrtf(1.0f + BN_EPS) : 0.0f;
#pragma unroll
    for (int r = 0; r < 8; ++r) {
        int oc = ocBase + r + (half ? 8 : 0);
        if (oc >= Cout) continue;
        float v = acc[r] + bias[oc];
        if (bnG) v = v * (bnG[oc] * bnS) + bnB[oc];
        if (relu) v = fmaxf(v, 0.0f);
        unsigned oidx = (unsigned)((((b * Cout + oc) * 128 + y) << 7) + xo);
        if (accum) out[oidx] += v; else out[oidx] = v;
    }
}

// ---------------------------------------------------------------------------
// Generic direct fp32 conv (small / bandwidth-bound layers), templated taps
// ---------------------------------------------------------------------------
template <int KH, int KW, int STRIDE, int PAD>
__global__ void conv_direct_kernel(const float* __restrict__ in, const float* __restrict__ w,
                                   const float* __restrict__ bias,
                                   const float* __restrict__ bnG, const float* __restrict__ bnB,
                                   float* __restrict__ out,
                                   int Cin, int inH, int inW, int Cout,
                                   int outH, int outW, int relu, int total)
{
    int i = blockIdx.x * blockDim.x + threadIdx.x;
    if (i >= total) return;
    int x = i % outW; int t = i / outW;
    int y = t % outH; t /= outH;
    int oc = t % Cout; int b = t / Cout;

    float acc = bias[oc];
    const float* wp = w + (size_t)oc * Cin * KH * KW;
    const float* ip = in + (size_t)b * Cin * inH * inW;
    for (int ci = 0; ci < Cin; ++ci) {
#pragma unroll
        for (int ky = 0; ky < KH; ++ky) {
            int iy = y * STRIDE + ky - PAD;
            if ((unsigned)iy >= (unsigned)inH) continue;
#pragma unroll
            for (int kx = 0; kx < KW; ++kx) {
                int ix = x * STRIDE + kx - PAD;
                if ((unsigned)ix >= (unsigned)inW) continue;
                acc += ip[(unsigned)((ci * inH + iy) * inW + ix)]
                     * wp[(unsigned)((ci * KH + ky) * KW + kx)];
            }
        }
    }
    if (bnG) acc = acc * (bnG[oc] * rsqrtf(1.0f + BN_EPS)) + bnB[oc];
    if (relu) acc = fmaxf(acc, 0.0f);
    out[i] = acc;
}

// ---------------------------------------------------------------------------
// blur broadcast: (16,1,8,8) scalars -> (16,64,128,128), with the reference's
// aliasing bug (flat[0,0] reads flat[0,1]). Writes channels [0,64) of E(80ch).
// ---------------------------------------------------------------------------
__global__ void blur_kernel(const float* __restrict__ t, float* __restrict__ E, int total)
{
    int i = blockIdx.x * blockDim.x + threadIdx.x;
    if (i >= total) return;                     // total = 16*64*16384
    int pix = i & 16383;
    int ch  = (i >> 14) & 63;
    int b   = i >> 20;
    int src = b * 64 + ch;
    if (b == 0 && ch == 0) src = 1;             // aliasing bug replication
    E[(unsigned)(((b * 80 + ch) << 14) + pix)] = t[src];
}

// concat: copy C(16,16,128,128) into channels [64,80) of E
__global__ void concat_kernel(const float* __restrict__ C, float* __restrict__ E, int total)
{
    int i = blockIdx.x * blockDim.x + threadIdx.x;
    if (i >= total) return;                     // total = 16*16*16384
    int pix = i & 16383;
    int c   = (i >> 14) & 15;
    int b   = i >> 18;
    E[(unsigned)(((b * 80 + 64 + c) << 14) + pix)] = C[i];
}

// pixel shuffle r=4: (16,16,128,128) -> (16,1,512,512)
__global__ void pixel_shuffle_kernel(const float* __restrict__ in, float* __restrict__ out, int total)
{
    int i = blockIdx.x * blockDim.x + threadIdx.x;
    if (i >= total) return;                     // total = 16*512*512
    int x2 = i & 511; int t = i >> 9;
    int y2 = t & 511; int b = t >> 9;
    int h = y2 >> 2, r1 = y2 & 3;
    int ww = x2 >> 2, r2 = x2 & 3;
    int ch = r1 * 4 + r2;
    out[i] = in[(unsigned)((((b * 16 + ch) * 128 + h) << 7) + ww)];
}

// final 1x1 conv (8->1) + sigmoid -> d_out (16,1,256,256)
__global__ void final_sigmoid_kernel(const float* __restrict__ in, const float* __restrict__ w,
                                     const float* __restrict__ bias, float* __restrict__ out, int total)
{
    int i = blockIdx.x * blockDim.x + threadIdx.x;
    if (i >= total) return;                     // total = 16*65536
    int pix = i & 65535;
    int b   = i >> 16;
    float acc = bias[0];
#pragma unroll
    for (int c = 0; c < 8; ++c)
        acc += in[(unsigned)(((b * 8 + c) << 16) + pix)] * w[c];
    out[i] = 1.0f / (1.0f + __expf(-acc));
}

// ---------------------------------------------------------------------------
// Host launch sequence
// ---------------------------------------------------------------------------
static inline dim3 g1(int total) { return dim3((total + TPB - 1) / TPB); }

extern "C" void kernel_launch(void* const* d_in, const int* in_sizes, int n_in,
                              void* d_out, int out_size, void* d_ws, size_t ws_size,
                              hipStream_t stream) {
    (void)in_sizes; (void)n_in; (void)out_size; (void)ws_size;
    const float* x = (const float*)d_in[0];
    #define P(i) ((const float*)d_in[i])

    float* ws = (float*)d_ws;
    // fp32 workspace arena (floats), with lifetime-based reuse. Peak ~228 MB.
    const size_t O_A  = 0;              // conv1 out  (16,8,256,256)  8,388,608
    const size_t O_B  = 8388608;        // conv2 out  (16,32,128,128) 8,388,608
    const size_t O_E  = 0;              // concat E   (16,80,128,128) 20,971,520 (A,B dead)
    const size_t O_C  = 20971520;       // conv3 out  (16,16,128,128) 4,194,304
    const size_t O_S2 = 25165824;       // F / H / M region           16,777,216
    const size_t O_S3 = 41943040;       // G / J region               7,864,320
    const size_t O_S4 = 49807360;       // I / K region               7,077,888
    const size_t O_D1 = O_S2;           // (16,4,64,64)
    const size_t O_D2 = O_S2 + 262144;  // (16,20,32,32)
    const size_t O_D3 = O_S2 + 589824;  // (16,30,16,16)
    const size_t O_D4 = O_S2 + 712704;  // (16,8,8,8)
    const size_t O_D5 = O_S2 + 720896;  // (16,1,8,8)

    // 1. conv1 + BN + ReLU : x(16,1,256,256) -> A(16,8,256,256)
    conv_direct_kernel<3,3,1,1><<<g1(8388608), TPB, 0, stream>>>(
        x, P(1), P(2), P(3), P(4), ws + O_A, 1, 256, 256, 8, 256, 256, 1, 8388608);

    // 2. conv2 (WMMA): A -> B(16,32,128,128), k5 s2 p2, ReLU
    wmma_conv_kernel<5,5,2,2><<<dim3(128, 2, 16), TPB, 0, stream>>>(
        ws + O_A, P(5), P(6), nullptr, nullptr, ws + O_B, 8, 256, 256, 32, 1, 0);

    // 3. conv3 (WMMA): B -> C(16,16,128,128), k3 s1 p1
    wmma_conv_kernel<3,3,1,1><<<dim3(128, 1, 16), TPB, 0, stream>>>(
        ws + O_B, P(7), P(8), nullptr, nullptr, ws + O_C, 32, 128, 128, 16, 0, 0);

    // 4-8. bottleneck chain (direct): conv4..conv8, all ReLU
    conv_direct_kernel<3,3,2,1><<<g1(262144), TPB, 0, stream>>>(
        ws + O_C, P(9), P(10), nullptr, nullptr, ws + O_D1, 16, 128, 128, 4, 64, 64, 1, 262144);
    conv_direct_kernel<3,3,2,1><<<g1(327680), TPB, 0, stream>>>(
        ws + O_D1, P(11), P(12), nullptr, nullptr, ws + O_D2, 4, 64, 64, 20, 32, 32, 1, 327680);
    conv_direct_kernel<5,5,2,2><<<g1(122880), TPB, 0, stream>>>(
        ws + O_D2, P(13), P(14), nullptr, nullptr, ws + O_D3, 20, 32, 32, 30, 16, 16, 1, 122880);
    conv_direct_kernel<3,3,2,1><<<g1(8192), TPB, 0, stream>>>(
        ws + O_D3, P(15), P(16), nullptr, nullptr, ws + O_D4, 30, 16, 16, 8, 8, 8, 1, 8192);
    conv_direct_kernel<1,1,1,0><<<g1(1024), TPB, 0, stream>>>(
        ws + O_D4, P(17), P(18), nullptr, nullptr, ws + O_D5, 8, 8, 8, 1, 8, 8, 1, 1024);

    // 9-10. blur broadcast into E[0:64) and concat C into E[64:80)
    blur_kernel<<<g1(16777216), TPB, 0, stream>>>(ws + O_D5, ws + O_E, 16777216);
    concat_kernel<<<g1(4194304), TPB, 0, stream>>>(ws + O_C, ws + O_E, 4194304);

    // 11. res1_conv1 (WMMA): E(80ch) -> F(64ch), BN + ReLU
    wmma_conv_kernel<3,3,1,1><<<dim3(128, 4, 16), TPB, 0, stream>>>(
        ws + O_E, P(19), P(20), P(21), P(22), ws + O_S2, 80, 128, 128, 64, 1, 0);
    // 12. res1_conv2 (WMMA): F -> G(20ch), BN
    wmma_conv_kernel<3,3,1,1><<<dim3(128, 2, 16), TPB, 0, stream>>>(
        ws + O_S2, P(23), P(24), P(25), P(26), ws + O_S3, 64, 128, 128, 20, 0, 0);
    // 13. conv1x1_1 (WMMA, accumulate): E(80ch) -> += G
    wmma_conv_kernel<1,1,1,0><<<dim3(128, 2, 16), TPB, 0, stream>>>(
        ws + O_E, P(27), P(28), nullptr, nullptr, ws + O_S3, 80, 128, 128, 20, 0, 1);

    // 14. res2_conv1 (WMMA): G -> H(50ch), BN + ReLU
    wmma_conv_kernel<3,3,1,1><<<dim3(128, 4, 16), TPB, 0, stream>>>(
        ws + O_S3, P(29), P(30), P(31), P(32), ws + O_S2, 20, 128, 128, 50, 1, 0);
    // 15. res2_conv2 (WMMA): H -> I(27ch), BN
    wmma_conv_kernel<3,3,1,1><<<dim3(128, 2, 16), TPB, 0, stream>>>(
        ws + O_S2, P(33), P(34), P(35), P(36), ws + O_S4, 50, 128, 128, 27, 0, 0);
    // 16. conv1x1_2 (WMMA, accumulate): G -> += I
    wmma_conv_kernel<1,1,1,0><<<dim3(128, 2, 16), TPB, 0, stream>>>(
        ws + O_S3, P(37), P(38), nullptr, nullptr, ws + O_S4, 20, 128, 128, 27, 0, 1);

    // 17. up_conv1 (WMMA): I -> J(30ch), ReLU
    wmma_conv_kernel<3,3,1,1><<<dim3(128, 2, 16), TPB, 0, stream>>>(
        ws + O_S4, P(39), P(40), nullptr, nullptr, ws + O_S3, 27, 128, 128, 30, 1, 0);
    // 18. up_conv2 (WMMA, 1x1): J -> K(16ch), ReLU
    wmma_conv_kernel<1,1,1,0><<<dim3(128, 1, 16), TPB, 0, stream>>>(
        ws + O_S3, P(41), P(42), nullptr, nullptr, ws + O_S4, 30, 128, 128, 16, 1, 0);

    // 19. pixel shuffle r=4: K(16,16,128,128) -> L(16,1,512,512) (reuse O_C)
    pixel_shuffle_kernel<<<g1(4194304), TPB, 0, stream>>>(ws + O_S4, ws + O_C, 4194304);

    // 20. up_conv3 (direct): L -> M(16,8,256,256), k3 s2 p1, ReLU (reuse O_S2)
    conv_direct_kernel<3,3,2,1><<<g1(8388608), TPB, 0, stream>>>(
        ws + O_C, P(43), P(44), nullptr, nullptr, ws + O_S2, 1, 512, 512, 8, 256, 256, 1, 8388608);

    // 21. up_conv4 1x1 + sigmoid -> d_out (16,1,256,256)
    final_sigmoid_kernel<<<g1(1048576), TPB, 0, stream>>>(
        ws + O_S2, P(45), P(46), (float*)d_out, 1048576);

    #undef P
}
